// Bn_bin_conv_pool_48309792145694
// MI455X (gfx1250) — matrix operand order repaired and checked by reading, not compiled, not run on hardware
//
#include <hip/hip_runtime.h>
#include <hip/hip_bf16.h>

typedef _Float16 v16h __attribute__((ext_vector_type(16)));
typedef _Float16 v8h  __attribute__((ext_vector_type(8)));
typedef float    v8f  __attribute__((ext_vector_type(8)));

#define CIN_   64
#define LIN_   4096
#define COUT_  128
#define KW_    7
#define PADW_  3
#define LOUT_  2048
#define LTILE_ 256              // conv output positions per block (pre-pool)
#define HALO_  6                // K-1
#define ROWS_  (LTILE_ + HALO_) // 262 input positions (incl. halo)
#define PITCH_ 72               // halfs per xT row (144B: 16B-aligned, bank-conflict-free)
#define NSTEPS_ 14              // K-dim = CIN*K = 448 -> 14 steps of 32
#define NTILES_ (LTILE_ / 16)   // 16 n-tiles per block
// xT size + slack for the (dead) cross-tile prefetches on the last n-tile:
#define XT_ELEMS_ (NTILES_ * 16 * PITCH_ + 16 * PITCH_ + 32)

__global__ __launch_bounds__(256)
void bn_bin_conv_pool_kernel(const float* __restrict__ I,
                             const float* __restrict__ bn_gamma,
                             const float* __restrict__ bn_beta,
                             const float* __restrict__ bn_mean,
                             const float* __restrict__ bn_var,
                             const float* __restrict__ conv_w,
                             const float* __restrict__ alpha,
                             const float* __restrict__ prelu_w,
                             float* __restrict__ out)
{
    __shared__ __align__(16) _Float16 xT[XT_ELEMS_]; // [pos][cin], ~39 KB
    __shared__ float bnScale[CIN_];
    __shared__ float bnShift[CIN_];

    const int tid    = threadIdx.x;
    const int b      = blockIdx.x >> 4;   // 16 L-tiles per batch
    const int lt     = blockIdx.x & 15;
    const int lstart = lt * LTILE_;

    // ---- Phase 0: fold BN into scale/shift per channel ----
    if (tid < CIN_) {
        float sc = bn_gamma[tid] * rsqrtf(bn_var[tid] + 1e-5f);
        bnScale[tid] = sc;
        bnShift[tid] = bn_beta[tid] - bn_mean[tid] * sc;
    }
    __syncthreads();

    // ---- Phase 1: BN + sign -> f16 tile in LDS, transposed to [pos][cin] ----
    const float* Ib = I + (size_t)b * CIN_ * LIN_;
    for (int i = tid; i < CIN_ * ROWS_; i += 256) {
        int cin = i / ROWS_;
        int pos = i - cin * ROWS_;
        int l   = lstart + pos - PADW_;
        _Float16 v = (_Float16)0.0f;             // zero padding binarizes to 0
        if (l >= 0 && l < LIN_) {
            float x = Ib[cin * LIN_ + l];
            float s = x * bnScale[cin] + bnShift[cin];
            float f = (s > 0.0f) ? 1.0f : ((s < 0.0f) ? -1.0f : 0.0f);
            v = (_Float16)f;
        }
        xT[pos * PITCH_ + cin] = v;
    }
    __syncthreads();

    // ---- Phase 2: implicit-GEMM with WMMA f16 ----
    const int lane = tid & 31;   // wave32
    const int wave = tid >> 5;   // 8 waves: wave w owns cout tile w (16 couts)
    const int n16  = lane & 15;
    const int hi   = lane >> 4;  // high half-wave

    // A fragments: weights in exact per-lane WMMA-A order, register resident.
    // K-dim flattening: kdim = kk*64 + cin ; step s: kk = s>>1, cin base = (s&1)*32.
    // A k order used here (contiguous, matches B): lane(0-15)=M; half j -> K = j + hi*16
    const int coutA = wave * 16 + n16;
    v16h afrag[NSTEPS_];
    #pragma unroll
    for (int s = 0; s < NSTEPS_; ++s) {
        const int kk = s >> 1;
        const int cb = (s & 1) * 32;
        #pragma unroll
        for (int j = 0; j < 16; ++j) {
            const int Kj  = j + hi * 16;
            const int cin = cb + Kj;
            afrag[s][j] = (_Float16)conv_w[(coutA * CIN_ + cin) * KW_ + kk];
        }
    }

    // Per-lane epilogue constants: C layout row r -> cout = wave*16 + r + 8*hi
    float alph[8];
    #pragma unroll
    for (int r = 0; r < 8; ++r) alph[r] = alpha[wave * 16 + r + 8 * hi];
    const float pw = prelu_w[0];

    float* Ob = out + (size_t)b * COUT_ * LOUT_;

    // Per-lane LDS base: row component n16, cin component hi*16
    const _Float16* lanep = &xT[n16 * PITCH_ + hi * 16];
    auto loadB = [&](int off) {
        union { v16h v; v8h h[2]; } bu;
        bu.h[0] = *(const v8h*)(lanep + off);
        bu.h[1] = *(const v8h*)(lanep + off + 8);
        return bu.v;
    };
    // element offset for step t within a tile (t may wrap into the next tile)
    auto poff = [](int t) {
        int add = 0;
        if (t >= NSTEPS_) { t -= NSTEPS_; add = 16 * PITCH_; }
        return add + (t >> 1) * PITCH_ + (t & 1) * 32;
    };

    // Ping-pong double buffer, prefetch distance 2: load(s+2); wmma(s);
    // load(s+3); wmma(s+1). Each wmma waits with ~4 DS loads outstanding.
    v16h b0 = loadB(poff(0));
    v16h b1 = loadB(poff(1));
    int ntoff = 0;

    #pragma unroll 1
    for (int nt = 0; nt < NTILES_; ++nt, ntoff += 16 * PITCH_) {
        v8f c = {};
        #pragma unroll
        for (int s = 0; s < NSTEPS_; s += 2) {
            v16h p0 = loadB(ntoff + poff(s + 2));   // prefetch for step s+2
            c = __builtin_amdgcn_wmma_f32_16x16x32_f16(
                    false, afrag[s], false, b0, (short)0, c, false, false);
            b0 = p0;
            v16h p1 = loadB(ntoff + poff(s + 3));   // prefetch for step s+3
            c = __builtin_amdgcn_wmma_f32_16x16x32_f16(
                    false, afrag[s + 1], false, b1, (short)0, c, false, false);
            b1 = p1;
        }
        // ---- epilogue: alpha, PReLU, maxpool(2) via lane-pair shuffle, store ----
        const int lloc = nt * 16 + n16;     // conv output position within tile
        #pragma unroll
        for (int r = 0; r < 8; ++r) {
            float y = c[r] * alph[r];
            y = (y > 0.0f) ? y : pw * y;
            float o = __shfl_xor(y, 1, 32); // partner conv position (2j <-> 2j+1)
            float m = fmaxf(y, o);
            if ((n16 & 1) == 0) {
                const int cout = wave * 16 + r + 8 * hi;
                const int ol   = (lstart + lloc) >> 1;
                Ob[cout * LOUT_ + ol] = m;
            }
        }
    }
}

extern "C" void kernel_launch(void* const* d_in, const int* in_sizes, int n_in,
                              void* d_out, int out_size, void* d_ws, size_t ws_size,
                              hipStream_t stream) {
    const float* I  = (const float*)d_in[0];
    const float* g  = (const float*)d_in[1];
    const float* be = (const float*)d_in[2];
    const float* mn = (const float*)d_in[3];
    const float* vr = (const float*)d_in[4];
    const float* w  = (const float*)d_in[5];
    const float* al = (const float*)d_in[6];
    const float* pr = (const float*)d_in[7];
    float* out = (float*)d_out;

    dim3 grid(64 * (LIN_ / LTILE_));   // B * L-tiles = 64 * 16 = 1024 blocks
    bn_bin_conv_pool_kernel<<<grid, 256, 0, stream>>>(I, g, be, mn, vr, w, al, pr, out);
}
